// CLGADN_39805756899947
// MI455X (gfx1250) — matrix operand order repaired and checked by reading, not compiled, not run on hardware
//
#include <hip/hip_runtime.h>
#include <hip/hip_bf16.h>
#include <math.h>

// ---------------- problem constants ----------------
#define NS    100000
#define NC    30000
#define NCAT  1000
#define DIM   100
#define BB    1024
#define SLEN  20
#define SS    21            // SL + 1
#define HH    4
#define DK    25            // DIM / H
#define EE    1000000
#define DFF   512
#define NTOT  130001        // NS + NC + 1
#define KP_D  128           // padded 100 -> 128 (mult of 32)
#define NP_D  112           // padded 100 -> 112 (mult of 16)
#define FEATK 2200          // S*D + D
#define FEATKP 2208         // padded, mult of 32

typedef __bf16 bf16;
typedef bf16  v16bf __attribute__((ext_vector_type(16)));
typedef bf16  v8bf  __attribute__((ext_vector_type(8)));
typedef float v8f   __attribute__((ext_vector_type(8)));

#define ACT_NONE  0
#define ACT_RELU  1
#define ACT_LEAKY 2

// ---------------- utility kernels ----------------
__global__ void zero_words_kernel(unsigned int* __restrict__ p, long n) {
  long i = (long)blockIdx.x * blockDim.x + threadIdx.x;
  if (i < n) p[i] = 0u;
}

__global__ void concat_init_kernel(const float* __restrict__ stu,
                                   const float* __restrict__ course,
                                   float* __restrict__ all0,
                                   float* __restrict__ mix) {
  long i = (long)blockIdx.x * blockDim.x + threadIdx.x;
  long n = (long)NTOT * DIM;
  if (i >= n) return;
  float v = (i < (long)NS * DIM) ? stu[i] : course[i - (long)NS * DIM];
  all0[i] = v;
  mix[i]  = v * (1.0f / 3.0f);   // e0 contributes 0.5/3 from each of the 2 graphs
}

// one wave per edge: lanes 0..24 each move a float4 (100 floats / row)
__global__ void gcn_scatter_kernel(const int* __restrict__ rows,
                                   const int* __restrict__ cols,
                                   const float* __restrict__ vals,
                                   const float* __restrict__ ein,
                                   float* __restrict__ eout, int nedges) {
  int wave  = blockIdx.x * (blockDim.x >> 5) + (threadIdx.x >> 5);
  int lane  = threadIdx.x & 31;
  int nwave = gridDim.x * (blockDim.x >> 5);
  for (int e = wave; e < nedges; e += nwave) {
    int r = rows[e], c = cols[e];
    float v = vals[e];
    if (lane < 25) {
      float4 s = *(const float4*)(ein + (size_t)c * DIM + lane * 4);
      float* d = eout + (size_t)r * DIM + lane * 4;
      atomicAdd(d + 0, v * s.x);
      atomicAdd(d + 1, v * s.y);
      atomicAdd(d + 2, v * s.z);
      atomicAdd(d + 3, v * s.w);
    }
  }
}

__global__ void accum_scaled_kernel(float* __restrict__ mix,
                                    const float* __restrict__ e, float scale) {
  long i = (long)blockIdx.x * blockDim.x + threadIdx.x;
  if (i < (long)NTOT * DIM) mix[i] += e[i] * scale;
}

// ---------------- sequence / gather kernels ----------------
__global__ void build_seq_kernel(const int* __restrict__ courses,
                                 const int* __restrict__ category,
                                 const int* __restrict__ candidate,
                                 const int* __restrict__ candidate_cate,
                                 const float* __restrict__ mix,
                                 const float* __restrict__ cat_emb,
                                 const float* __restrict__ pos_emb,
                                 float* __restrict__ seq,
                                 bf16* __restrict__ seq_bf) {
  int rs = blockIdx.x;            // b*S + s
  int b = rs / SS, s = rs % SS;
  int d = threadIdx.x;            // 0..127
  int c, cat;
  if (s < SLEN) { c = courses[b * SLEN + s]; cat = category[b * SLEN + s]; }
  else          { c = candidate[b];          cat = candidate_cate[b]; }
  if (d < DIM) {
    float v = mix[(size_t)(NS + c) * DIM + d] + cat_emb[(size_t)cat * DIM + d]
            + pos_emb[s * DIM + d];
    seq[(size_t)rs * DIM + d] = v;
    seq_bf[(size_t)rs * KP_D + d] = (bf16)v;
  } else {
    seq_bf[(size_t)rs * KP_D + d] = (bf16)0.f;
  }
}

__global__ void gather_u_kernel(const int* __restrict__ student,
                                const float* __restrict__ mix,
                                float* __restrict__ u) {
  int b = blockIdx.x, d = threadIdx.x;
  if (d < DIM) u[(size_t)b * DIM + d] = mix[(size_t)student[b] * DIM + d];
}

// W [K x N] f32 row-major  ->  Bt [Npad x Kp] bf16 (column-major weight, zero pad)
__global__ void convert_weight_kernel(const float* __restrict__ W,
                                      bf16* __restrict__ Bt,
                                      int K, int N, int Kp, int Npad) {
  long idx = (long)blockIdx.x * blockDim.x + threadIdx.x;
  long total = (long)Npad * Kp;
  if (idx >= total) return;
  int n = (int)(idx / Kp), k = (int)(idx % Kp);
  float v = (n < N && k < K) ? W[(size_t)k * N + n] : 0.f;
  Bt[idx] = (bf16)v;
}

__global__ void cast_f32_bf16_kernel(const float* __restrict__ s,
                                     bf16* __restrict__ d, long n) {
  long i = (long)blockIdx.x * blockDim.x + threadIdx.x;
  if (i < n) d[i] = (bf16)s[i];
}

// ---------------- WMMA GEMM:  C[M x Nreal] = act(A[Mtiles*16 x Kp] * Bt^T + bias)
// A row-major bf16 (stride Kp), Bt is [Npad x Kp] bf16 (weight column n contiguous in K)
__global__ void gemm_bf16_wmma_kernel(const bf16* __restrict__ A,
                                      const bf16* __restrict__ Bt,
                                      const float* __restrict__ bias,
                                      float* __restrict__ C,
                                      int Kp, int Nreal, int ldc,
                                      int Mtiles, int act) {
  int nTile = blockIdx.x;
  int mTile = blockIdx.y * blockDim.y + threadIdx.y;
  if (mTile >= Mtiles) return;                   // wave-uniform: EXEC stays full
  int lane = threadIdx.x;                        // 0..31
  int lm = lane & 15;
  int hi = lane >> 4;                            // 0 or 1
  const bf16* arow = A  + (size_t)(mTile * 16 + lm) * Kp + hi * 16;
  const bf16* brow = Bt + (size_t)(nTile * 16 + lm) * Kp + hi * 16;
  v8f acc = {};
  for (int k0 = 0; k0 < Kp; k0 += 32) {
    // A 16x32 bf16 fragment: lanes0-15 K{0-15}, lanes16-31 K{16-31}
    v8bf alo = *(const v8bf*)(arow + k0);
    v8bf ahi = *(const v8bf*)(arow + k0 + 8);
    v16bf a = __builtin_shufflevector(alo, ahi,
        0,1,2,3,4,5,6,7,8,9,10,11,12,13,14,15);
    // B 32x16 bf16 fragment: lane = n, K half picked by lane[4]; contiguous in Bt
    v16bf b = *(const v16bf*)(brow + k0);
    acc = __builtin_amdgcn_wmma_f32_16x16x32_bf16(
        false, a, false, b, (short)0, acc, false, false);
  }
  int n = nTile * 16 + lm;
  if (n >= Nreal) return;
  float bv = bias ? bias[n] : 0.f;
#pragma unroll
  for (int r = 0; r < 8; ++r) {
    int row = mTile * 16 + r + hi * 8;           // D layout: VGPR r -> M=r+8*hi
    float x = acc[r] + bv;
    if (act == ACT_RELU)       x = fmaxf(x, 0.f);
    else if (act == ACT_LEAKY) x = (x > 0.f) ? x : 0.01f * x;
    C[(size_t)row * ldc + n] = x;
  }
}

// ---------------- attention core (21x21 per (b,h)) ----------------
__global__ void attention_kernel(const float* __restrict__ q,
                                 const float* __restrict__ v,
                                 const float* __restrict__ gammas,
                                 bf16* __restrict__ ctx_bf) {
  int t = blockIdx.x * blockDim.x + threadIdx.x;
  if (t >= BB * HH * SS) return;
  int i = t % SS;
  int h = (t / SS) % HH;
  int b = t / (SS * HH);
  const float* qb = q + (size_t)b * SS * DIM + h * DK;
  float g  = gammas[h];
  float sp = (g > 20.f) ? g : log1pf(expf(g));   // softplus; gamma = -sp
  float sc[SS];
  float mx = -1e30f;
  for (int j = 0; j <= i; ++j) {
    float dot = 0.f;
    for (int dk = 0; dk < DK; ++dk)
      dot += qb[(size_t)i * DIM + dk] * qb[(size_t)j * DIM + dk];
    float eff = expf(((float)(i - j) / 21.f) * sp);     // exp(-pos_eff*gamma)
    eff = fminf(fmaxf(eff, 1e-5f), 1e5f);
    float s = dot * 0.2f * eff;                          // / sqrt(DK)=5
    sc[j] = s;
    mx = fmaxf(mx, s);
  }
  float sum = 0.f;
  for (int j = 0; j <= i; ++j) { sc[j] = expf(sc[j] - mx); sum += sc[j]; }
  float inv = 1.f / sum;
  const float* vb = v + (size_t)b * SS * DIM + h * DK;
  bf16* cb = ctx_bf + (size_t)(b * SS + i) * KP_D + h * DK;
  for (int dk = 0; dk < DK; ++dk) {
    float acc = 0.f;
    for (int j = 0; j <= i; ++j) acc += sc[j] * vb[(size_t)j * DIM + dk];
    cb[dk] = (bf16)(acc * inv);
  }
}

// ---------------- residual + LayerNorm, one wave32 per row ----------------
__global__ void add_ln_kernel(const float* __restrict__ a,
                              const float* __restrict__ r,
                              const float* __restrict__ g,
                              const float* __restrict__ be,
                              float* __restrict__ out,
                              bf16* __restrict__ out_bf, int nrows) {
  int wave = blockIdx.x * (blockDim.x >> 5) + (threadIdx.x >> 5);
  int lane = threadIdx.x & 31;
  if (wave >= nrows) return;
  const float* ra = a + (size_t)wave * DIM;
  const float* rb = r + (size_t)wave * DIM;
  float vals[4];
  float s = 0.f;
#pragma unroll
  for (int tt = 0; tt < 4; ++tt) {
    int d = lane + tt * 32;
    float x = (d < DIM) ? (ra[d] + rb[d]) : 0.f;
    vals[tt] = x; s += x;
  }
  for (int o = 16; o > 0; o >>= 1) s += __shfl_xor(s, o, 32);
  float mu = s * (1.0f / DIM);
  float vv = 0.f;
#pragma unroll
  for (int tt = 0; tt < 4; ++tt) {
    int d = lane + tt * 32;
    if (d < DIM) { float dd = vals[tt] - mu; vv += dd * dd; }
  }
  for (int o = 16; o > 0; o >>= 1) vv += __shfl_xor(vv, o, 32);
  float rstd = rsqrtf(vv * (1.0f / DIM) + 1e-5f);
#pragma unroll
  for (int tt = 0; tt < 4; ++tt) {
    int d = lane + tt * 32;
    if (d < DIM) {
      float y = g[d] * (vals[tt] - mu) * rstd + be[d];
      out[(size_t)wave * DIM + d] = y;
      if (out_bf) out_bf[(size_t)wave * KP_D + d] = (bf16)y;
    } else if (out_bf) {
      out_bf[(size_t)wave * KP_D + d] = (bf16)0.f;
    }
  }
}

__global__ void build_feat_kernel(const float* __restrict__ x,
                                  const float* __restrict__ u,
                                  bf16* __restrict__ feat) {
  long idx = (long)blockIdx.x * blockDim.x + threadIdx.x;
  if (idx >= (long)BB * FEATKP) return;
  int b = (int)(idx / FEATKP), k = (int)(idx % FEATKP);
  float v;
  if (k < SS * DIM)        v = x[(size_t)b * SS * DIM + k];
  else if (k < FEATK)      v = u[(size_t)b * DIM + (k - SS * DIM)];
  else                     v = 0.f;
  feat[idx] = (bf16)v;
}

__global__ void final_kernel(const float* __restrict__ h3,
                             const float* __restrict__ M4,
                             const float* __restrict__ mb4,
                             float* __restrict__ out) {
  int b = blockIdx.x * blockDim.x + threadIdx.x;
  if (b >= BB) return;
  float acc = mb4[0];
  for (int k = 0; k < 256; ++k) acc += h3[(size_t)b * 256 + k] * M4[k];
  out[b] = 1.f / (1.f + expf(-acc));
}

// ---------------- host orchestration ----------------
static inline size_t a256(size_t x) { return (x + 255) & ~(size_t)255; }
static inline int cdiv(long a, long b) { return (int)((a + b - 1) / b); }

extern "C" void kernel_launch(void* const* d_in, const int* in_sizes, int n_in,
                              void* d_out, int out_size, void* d_ws, size_t ws_size,
                              hipStream_t stream) {
  const int*   student        = (const int*)d_in[0];
  const int*   courses        = (const int*)d_in[1];
  const int*   category       = (const int*)d_in[2];
  const int*   candidate      = (const int*)d_in[3];
  const int*   candidate_cate = (const int*)d_in[4];
  const int*   rows1 = (const int*)d_in[5];
  const int*   cols1 = (const int*)d_in[6];
  const float* vals1 = (const float*)d_in[7];
  const int*   rows2 = (const int*)d_in[8];
  const int*   cols2 = (const int*)d_in[9];
  const float* vals2 = (const float*)d_in[10];
  const float* stu_emb    = (const float*)d_in[11];
  const float* course_emb = (const float*)d_in[12];
  const float* cat_emb    = (const float*)d_in[13];
  const float* pos_emb    = (const float*)d_in[14];
  const float* Wk = (const float*)d_in[15]; const float* bk = (const float*)d_in[16];
  const float* Wv = (const float*)d_in[17]; const float* bv = (const float*)d_in[18];
  const float* Wo = (const float*)d_in[19]; const float* bo = (const float*)d_in[20];
  const float* gammas = (const float*)d_in[21];
  const float* ln1_g = (const float*)d_in[22]; const float* ln1_b = (const float*)d_in[23];
  const float* W1 = (const float*)d_in[24]; const float* b1 = (const float*)d_in[25];
  const float* W2 = (const float*)d_in[26]; const float* b2 = (const float*)d_in[27];
  const float* ln2_g = (const float*)d_in[28]; const float* ln2_b = (const float*)d_in[29];
  const float* M1 = (const float*)d_in[30]; const float* mb1 = (const float*)d_in[31];
  const float* M2 = (const float*)d_in[32]; const float* mb2 = (const float*)d_in[33];
  const float* M3 = (const float*)d_in[34]; const float* mb3 = (const float*)d_in[35];
  const float* M4 = (const float*)d_in[36]; const float* mb4 = (const float*)d_in[37];
  float* out = (float*)d_out;

  char* base = (char*)d_ws;
  size_t off = 0;
  auto carve = [&](size_t bytes) -> char* { char* p = base + off; off += a256(bytes); return p; };

  const long ND = (long)NTOT * DIM;           // 13,000,100
  const int  MROWS = BB * SS;                 // 21504 (mult of 16)
  const int  MT = MROWS / 16;                 // 1344

  float* mix = (float*)carve(ND * 4);
  size_t mark = off;                          // phase-1 region (reused in phase 2)

  // ---- phase 1: LightGCN ----
  float* all0 = (float*)carve(ND * 4);
  float* eA   = (float*)carve(ND * 4);
  float* eB   = (float*)carve(ND * 4);

  concat_init_kernel<<<cdiv(ND, 256), 256, 0, stream>>>(stu_emb, course_emb, all0, mix);

  const int* grows[2] = { rows1, rows2 };
  const int* gcols[2] = { cols1, cols2 };
  const float* gvals[2] = { vals1, vals2 };
  for (int g = 0; g < 2; ++g) {
    zero_words_kernel<<<cdiv(ND, 256), 256, 0, stream>>>((unsigned int*)eA, ND);
    gcn_scatter_kernel<<<2048, 256, 0, stream>>>(grows[g], gcols[g], gvals[g], all0, eA, EE);
    accum_scaled_kernel<<<cdiv(ND, 256), 256, 0, stream>>>(mix, eA, 0.5f / 3.0f);
    zero_words_kernel<<<cdiv(ND, 256), 256, 0, stream>>>((unsigned int*)eB, ND);
    gcn_scatter_kernel<<<2048, 256, 0, stream>>>(grows[g], gcols[g], gvals[g], eA, eB, EE);
    accum_scaled_kernel<<<cdiv(ND, 256), 256, 0, stream>>>(mix, eB, 0.5f / 3.0f);
  }

  // ---- phase 2: reuse phase-1 scratch ----
  off = mark;
  float* seq      = (float*)carve((size_t)MROWS * DIM * 4);
  bf16*  seq_bf   = (bf16*) carve((size_t)MROWS * KP_D * 2);
  float* qbuf     = (float*)carve((size_t)MROWS * DIM * 4);
  float* vbuf     = (float*)carve((size_t)MROWS * DIM * 4);
  bf16*  ctx_bf   = (bf16*) carve((size_t)MROWS * KP_D * 2);
  float* attn_out = (float*)carve((size_t)MROWS * DIM * 4);
  float* x1       = (float*)carve((size_t)MROWS * DIM * 4);
  bf16*  x1_bf    = (bf16*) carve((size_t)MROWS * KP_D * 2);
  float* ffn1     = (float*)carve((size_t)MROWS * DFF * 4);
  bf16*  ffn1_bf  = (bf16*) carve((size_t)MROWS * DFF * 2);
  float* ffn2     = (float*)carve((size_t)MROWS * DIM * 4);
  float* xout     = (float*)carve((size_t)MROWS * DIM * 4);
  float* ubuf     = (float*)carve((size_t)BB * DIM * 4);
  bf16*  feat_bf  = (bf16*) carve((size_t)BB * FEATKP * 2);
  float* h1       = (float*)carve((size_t)BB * 1024 * 4);
  bf16*  h1_bf    = (bf16*) carve((size_t)BB * 1024 * 2);
  float* h2       = (float*)carve((size_t)BB * 512 * 4);
  bf16*  h2_bf    = (bf16*) carve((size_t)BB * 512 * 2);
  float* h3       = (float*)carve((size_t)BB * 256 * 4);
  bf16*  Wk_bt = (bf16*)carve((size_t)NP_D * KP_D * 2);
  bf16*  Wv_bt = (bf16*)carve((size_t)NP_D * KP_D * 2);
  bf16*  Wo_bt = (bf16*)carve((size_t)NP_D * KP_D * 2);
  bf16*  W1_bt = (bf16*)carve((size_t)DFF * KP_D * 2);
  bf16*  W2_bt = (bf16*)carve((size_t)NP_D * DFF * 2);
  bf16*  M1_bt = (bf16*)carve((size_t)1024 * FEATKP * 2);
  bf16*  M2_bt = (bf16*)carve((size_t)512 * 1024 * 2);
  bf16*  M3_bt = (bf16*)carve((size_t)256 * 512 * 2);
  bf16*  M4_bt = (bf16*)carve((size_t)16 * 256 * 2);

  // weight transpose+pad+cast
  auto cw = [&](const float* W, bf16* Bt, int K, int N, int Kp, int Npad) {
    long tot = (long)Npad * Kp;
    convert_weight_kernel<<<cdiv(tot, 256), 256, 0, stream>>>(W, Bt, K, N, Kp, Npad);
  };
  cw(Wk, Wk_bt, DIM, DIM, KP_D, NP_D);
  cw(Wv, Wv_bt, DIM, DIM, KP_D, NP_D);
  cw(Wo, Wo_bt, DIM, DIM, KP_D, NP_D);
  cw(W1, W1_bt, DIM, DFF, KP_D, DFF);
  cw(W2, W2_bt, DFF, DIM, DFF, NP_D);
  cw(M1, M1_bt, FEATK, 1024, FEATKP, 1024);
  cw(M2, M2_bt, 1024, 512, 1024, 512);
  cw(M3, M3_bt, 512, 256, 512, 256);
  cw(M4, M4_bt, 256, 1, 256, 16);

  // sequence + user gathers
  build_seq_kernel<<<MROWS, 128, 0, stream>>>(courses, category, candidate,
                                              candidate_cate, mix, cat_emb,
                                              pos_emb, seq, seq_bf);
  gather_u_kernel<<<BB, 128, 0, stream>>>(student, mix, ubuf);

  auto gemm = [&](const bf16* A, const bf16* Bt, const float* bias, float* C,
                  int Kp, int Nreal, int ldc, int Mtiles, int Ntiles, int act) {
    dim3 blk(32, 4);
    dim3 grd(Ntiles, cdiv(Mtiles, 4));
    gemm_bf16_wmma_kernel<<<grd, blk, 0, stream>>>(A, Bt, bias, C, Kp, Nreal,
                                                   ldc, Mtiles, act);
  };

  // q = seq@Wk+bk ; v = seq@Wv+bv
  gemm(seq_bf, Wk_bt, bk, qbuf, KP_D, DIM, DIM, MT, NP_D / 16, ACT_NONE);
  gemm(seq_bf, Wv_bt, bv, vbuf, KP_D, DIM, DIM, MT, NP_D / 16, ACT_NONE);

  // attention -> ctx (bf16, padded to 128 cols)
  zero_words_kernel<<<cdiv((long)MROWS * KP_D / 2, 256), 256, 0, stream>>>(
      (unsigned int*)ctx_bf, (long)MROWS * KP_D / 2);
  attention_kernel<<<cdiv((long)BB * HH * SS, 256), 256, 0, stream>>>(
      qbuf, vbuf, gammas, ctx_bf);

  // attn_out = ctx@Wo+bo ; x1 = LN(seq + attn_out)
  gemm(ctx_bf, Wo_bt, bo, attn_out, KP_D, DIM, DIM, MT, NP_D / 16, ACT_NONE);
  add_ln_kernel<<<cdiv(MROWS, 8), 256, 0, stream>>>(seq, attn_out, ln1_g, ln1_b,
                                                    x1, x1_bf, MROWS);

  // ffn = relu(x1@W1+b1)@W2+b2 ; x = LN(x1 + ffn)
  gemm(x1_bf, W1_bt, b1, ffn1, KP_D, DFF, DFF, MT, DFF / 16, ACT_RELU);
  cast_f32_bf16_kernel<<<cdiv((long)MROWS * DFF, 256), 256, 0, stream>>>(
      ffn1, ffn1_bf, (long)MROWS * DFF);
  gemm(ffn1_bf, W2_bt, b2, ffn2, DFF, DIM, DIM, MT, NP_D / 16, ACT_NONE);
  add_ln_kernel<<<cdiv(MROWS, 8), 256, 0, stream>>>(x1, ffn2, ln2_g, ln2_b,
                                                    xout, (bf16*)nullptr, MROWS);

  // feat = [x.reshape(B, S*D), u]  (bf16, padded K)
  build_feat_kernel<<<cdiv((long)BB * FEATKP, 256), 256, 0, stream>>>(xout, ubuf, feat_bf);

  // MLP head
  gemm(feat_bf, M1_bt, mb1, h1, FEATKP, 1024, 1024, BB / 16, 1024 / 16, ACT_LEAKY);
  cast_f32_bf16_kernel<<<cdiv((long)BB * 1024, 256), 256, 0, stream>>>(h1, h1_bf, (long)BB * 1024);
  gemm(h1_bf, M2_bt, mb2, h2, 1024, 512, 512, BB / 16, 512 / 16, ACT_LEAKY);
  cast_f32_bf16_kernel<<<cdiv((long)BB * 512, 256), 256, 0, stream>>>(h2, h2_bf, (long)BB * 512);
  gemm(h2_bf, M3_bt, mb3, h3, 512, 256, 256, BB / 16, 256 / 16, ACT_LEAKY);

  final_kernel<<<cdiv(BB, 256), 256, 0, stream>>>(h3, M4, mb4, out);

  (void)in_sizes; (void)n_in; (void)out_size; (void)ws_size;
}
